// Model_75015898792074
// MI455X (gfx1250) — matrix operand order repaired and checked by reading, not compiled, not run on hardware
//
#include <hip/hip_runtime.h>
#include <math.h>

#define D_MODELc 1024
#define D_FFc    4096
#define SEQc     192
#define BATCHc   8
#define BLc      (BATCHc*SEQc)   // 1536
#define NFEATc   7
#define NMARKc   4
#define KTOPc    15
#define HALFSEQc 96
#define MAVGPAD  12              // (25-1)/2

typedef __bf16 bf16_t;
typedef bf16_t v16bf __attribute__((ext_vector_type(16)));
typedef float  v8f   __attribute__((ext_vector_type(8)));

__device__ __forceinline__ float gelu_f(float x){
  float x3 = x*x*x;
  return 0.5f*x*(1.0f + tanhf(0.7978845608028654f*(x + 0.044715f*x3)));
}

// ---------------------------------------------------------------------------
// WMMA bf16 GEMM:  C[M,N] = act( A[M,K] * B[K,N] + bias ) + resid
// A,B,C,resid f32 row-major; converted to bf16 in LDS; f32 WMMA accumulate.
// Block tile 128x128, 256 threads = 8 waves (4x2), wave tile 32x64.
// ---------------------------------------------------------------------------
#define TM 128
#define TN 128
#define TK 32
#define LDP 8   // LDS row padding (bf16 elems)

__global__ __launch_bounds__(256) void gemm_wmma_kernel(
    const float* __restrict__ A, const float* __restrict__ B,
    const float* __restrict__ bias, const float* __restrict__ resid,
    float* __restrict__ C, int M, int N, int K, int act)
{
  __shared__ bf16_t sA[TM][TK + LDP];   // [m][k]
  __shared__ bf16_t sB[TN][TK + LDP];   // [n][k]  (B stored transposed)

  const int bm   = blockIdx.y * TM;
  const int bn   = blockIdx.x * TN;
  const int tid  = threadIdx.x;
  const int lane = tid & 31;
  const int wave = tid >> 5;
  const int wm   = (wave & 3) * 32;
  const int wn   = (wave >> 2) * 64;
  const int half = lane >> 4;
  const int lrow = lane & 15;

  v8f acc[2][4];
  const v8f vzero = {0.f,0.f,0.f,0.f,0.f,0.f,0.f,0.f};
#pragma unroll
  for (int i = 0; i < 2; ++i)
#pragma unroll
    for (int j = 0; j < 4; ++j) acc[i][j] = vzero;

  for (int k0 = 0; k0 < K; k0 += TK) {
    // ---- stage A tile (128x32): 2 threads per row, 16 f32 each
    {
      const int row  = tid >> 1;
      const int colb = (tid & 1) * 16;
      const float* src = A + (size_t)(bm + row) * K + (k0 + colb);
      if (k0 + TK < K) __builtin_prefetch(src + TK, 0, 1);
#pragma unroll
      for (int i = 0; i < 16; ++i) sA[row][colb + i] = (bf16_t)src[i];
    }
    // ---- stage B tile (32x128) transposed into sB[n][k]
    {
      const int kr = tid >> 3;
      const int cb = (tid & 7) * 16;
      const float* src = B + (size_t)(k0 + kr) * N + (bn + cb);
      if (k0 + TK < K) __builtin_prefetch(src + (size_t)TK * N, 0, 1);
#pragma unroll
      for (int i = 0; i < 16; ++i) sB[cb + i][kr] = (bf16_t)src[i];
    }
    __syncthreads();

    // ---- load fragments (contiguous k = 16*half + s, same order in A and B)
    v16bf af[2];
#pragma unroll
    for (int mi = 0; mi < 2; ++mi) {
      const bf16_t* base = &sA[wm + mi*16 + lrow][0];
#pragma unroll
      for (int t = 0; t < 8; ++t) {
        const int kk = half*16 + t*2;
        af[mi][2*t]   = base[kk];
        af[mi][2*t+1] = base[kk+1];
      }
    }
    v16bf bfr[4];
#pragma unroll
    for (int ni = 0; ni < 4; ++ni) {
      const bf16_t* base = &sB[wn + ni*16 + lrow][16*half];
#pragma unroll
      for (int t = 0; t < 16; ++t) bfr[ni][t] = base[t];
    }
#pragma unroll
    for (int mi = 0; mi < 2; ++mi)
#pragma unroll
      for (int ni = 0; ni < 4; ++ni)
        acc[mi][ni] = __builtin_amdgcn_wmma_f32_16x16x32_bf16(
            false, af[mi], false, bfr[ni], (short)0, acc[mi][ni], false, false);
    __syncthreads();
  }

  // ---- epilogue: C/D layout -> row = base + v + 8*half, col = base + lane%16
#pragma unroll
  for (int mi = 0; mi < 2; ++mi)
#pragma unroll
    for (int ni = 0; ni < 4; ++ni) {
      const int gc = bn + wn + ni*16 + lrow;
      const float bv = bias ? bias[gc] : 0.f;
#pragma unroll
      for (int v = 0; v < 8; ++v) {
        const int gr = bm + wm + mi*16 + v + 8*half;
        float val = acc[mi][ni][v] + bv;
        if (act == 1) val = gelu_f(val);
        const size_t off = (size_t)gr * N + gc;
        if (resid) val += resid[off];
        C[off] = val;
      }
    }
}

// ---------------------------------------------------------------------------
// Embedding: circular conv3 over 7 features + mark @ temp_w
// ---------------------------------------------------------------------------
__global__ void embed_kernel(const float* __restrict__ x, const float* __restrict__ mark,
                             const float* __restrict__ cw, const float* __restrict__ tw,
                             float* __restrict__ out)
{
  const int i = blockIdx.x * blockDim.x + threadIdx.x;
  if (i >= BLc * D_MODELc) return;
  const int o = i & (D_MODELc - 1);
  const int l = (i >> 10) % SEQc;
  const int b = i / (D_MODELc * SEQc);
  const int lm = (l + SEQc - 1) % SEQc, lp = (l + 1) % SEQc;
  const float* xm = x + (size_t)(b*SEQc + lm)*NFEATc;
  const float* x0 = x + (size_t)(b*SEQc + l )*NFEATc;
  const float* xp = x + (size_t)(b*SEQc + lp)*NFEATc;
  float s = 0.f;
#pragma unroll
  for (int c = 0; c < NFEATc; ++c) {
    const float* w = cw + o*(NFEATc*3) + c*3;
    s += xm[c]*w[0] + x0[c]*w[1] + xp[c]*w[2];
  }
  const float* mk = mark + (size_t)(b*SEQc + l)*NMARKc;
#pragma unroll
  for (int m = 0; m < NMARKc; ++m) s += mk[m] * tw[m*D_MODELc + o];
  out[i] = s;
}

// ---------------------------------------------------------------------------
// Series decomposition: moving mean k=25, replicate pad == clamped window.
// trend_mode: 0 = ignore trend, 1 = store, 2 = accumulate
// ---------------------------------------------------------------------------
__global__ void decomp_kernel(const float* __restrict__ x, float* __restrict__ seasonal,
                              float* __restrict__ trend, int C, int trend_mode)
{
  const int i = blockIdx.x * blockDim.x + threadIdx.x;
  if (i >= BATCHc * SEQc * C) return;
  const int c = i % C;
  const int l = (i / C) % SEQc;
  const int b = i / (C * SEQc);
  const float* xb = x + (size_t)b * SEQc * C + c;
  float s = 0.f;
#pragma unroll 5
  for (int j = -MAVGPAD; j <= MAVGPAD; ++j) {
    int lj = l + j;
    lj = lj < 0 ? 0 : (lj > SEQc-1 ? SEQc-1 : lj);
    s += xb[(size_t)lj * C];
  }
  const float m = s * (1.f / 25.f);
  seasonal[i] = x[i] - m;
  if (trend_mode == 1) trend[i] = m;
  else if (trend_mode == 2) trend[i] += m;
}

// ---------------------------------------------------------------------------
// Autocorrelation mean: corr[b,tau] = (1/D) sum_l q[b,l,:].k[b,(l+tau)%L,:]
// one block per (tau, b)
// ---------------------------------------------------------------------------
__global__ __launch_bounds__(256) void corr_kernel(const float* __restrict__ q,
                                                   const float* __restrict__ k,
                                                   float* __restrict__ corr)
{
  __shared__ float red[256];
  const int tau = blockIdx.x, b = blockIdx.y;
  const int tid = threadIdx.x;
  const float* qb = q + (size_t)b * SEQc * D_MODELc;
  const float* kb = k + (size_t)b * SEQc * D_MODELc;
  float s = 0.f;
  for (int l = 0; l < SEQc; ++l) {
    int lk = l + tau; if (lk >= SEQc) lk -= SEQc;
    const float* qr = qb + (size_t)l  * D_MODELc;
    const float* kr = kb + (size_t)lk * D_MODELc;
#pragma unroll
    for (int d = tid; d < D_MODELc; d += 256) s += qr[d] * kr[d];
  }
  red[tid] = s; __syncthreads();
  for (int st = 128; st > 0; st >>= 1) {
    if (tid < st) red[tid] += red[tid + st];
    __syncthreads();
  }
  if (tid == 0) corr[b * SEQc + tau] = red[0] * (1.f / (float)D_MODELc);
}

// top-15 + softmax, one block per batch
__global__ void topk_kernel(const float* __restrict__ corr, int* __restrict__ delay,
                            float* __restrict__ tc)
{
  __shared__ float sc[SEQc];
  const int b = blockIdx.x;
  for (int i = threadIdx.x; i < SEQc; i += blockDim.x) sc[i] = corr[b*SEQc + i];
  __syncthreads();
  if (threadIdx.x != 0) return;
  float w[KTOPc]; int idx[KTOPc];
  for (int i = 0; i < KTOPc; ++i) {
    float best = -1e30f; int bi = 0;
    for (int j = 0; j < SEQc; ++j) if (sc[j] > best) { best = sc[j]; bi = j; }
    w[i] = best; idx[i] = bi; sc[bi] = -1e30f;
  }
  const float mx = w[0];
  float sum = 0.f;
  for (int i = 0; i < KTOPc; ++i) { w[i] = __expf(w[i] - mx); sum += w[i]; }
  const float inv = 1.f / sum;
  for (int i = 0; i < KTOPc; ++i) { tc[b*16 + i] = w[i] * inv; delay[b*16 + i] = idx[i]; }
}

// out[b,l,d] = sum_i tc[b,i] * v[b,(l+delay[b,i])%L,d]
__global__ void gather_kernel(const float* __restrict__ v, const int* __restrict__ delay,
                              const float* __restrict__ tc, float* __restrict__ out)
{
  const int i = blockIdx.x * blockDim.x + threadIdx.x;
  if (i >= BLc * D_MODELc) return;
  const int d = i & (D_MODELc - 1);
  const int l = (i >> 10) % SEQc;
  const int b = i / (D_MODELc * SEQc);
  const float* vb = v + (size_t)b * SEQc * D_MODELc + d;
  float s = 0.f;
#pragma unroll
  for (int kk = 0; kk < KTOPc; ++kk) {
    int ls = l + delay[b*16 + kk]; if (ls >= SEQc) ls -= SEQc;
    s += tc[b*16 + kk] * vb[(size_t)ls * D_MODELc];
  }
  out[i] = s;
}

// ---------------------------------------------------------------------------
// my_layernorm phase1: per-row LN over D; phase2: subtract per-(b,d) seq mean
// ---------------------------------------------------------------------------
__global__ __launch_bounds__(256) void ln1_kernel(const float* __restrict__ x,
                                                  float* __restrict__ xh,
                                                  const float* __restrict__ g,
                                                  const float* __restrict__ bta)
{
  __shared__ float rs_[256], rq_[256];
  const int row = blockIdx.x;
  const int tid = threadIdx.x;
  const float* xr = x + (size_t)row * D_MODELc;
  float s = 0.f, s2 = 0.f;
#pragma unroll
  for (int d = tid; d < D_MODELc; d += 256) { float v = xr[d]; s += v; s2 += v*v; }
  rs_[tid] = s; rq_[tid] = s2; __syncthreads();
  for (int st = 128; st > 0; st >>= 1) {
    if (tid < st) { rs_[tid] += rs_[tid+st]; rq_[tid] += rq_[tid+st]; }
    __syncthreads();
  }
  const float mu  = rs_[0] * (1.f / D_MODELc);
  const float var = rq_[0] * (1.f / D_MODELc) - mu*mu;
  const float rsq = rsqrtf(var + 1e-5f);
#pragma unroll
  for (int d = tid; d < D_MODELc; d += 256)
    xh[(size_t)row * D_MODELc + d] = (xr[d] - mu) * rsq * g[d] + bta[d];
}

__global__ void ln2_kernel(const float* __restrict__ xh, float* __restrict__ out)
{
  const int i = blockIdx.x * blockDim.x + threadIdx.x;
  if (i >= BATCHc * D_MODELc) return;
  const int d = i & (D_MODELc - 1);
  const int b = i >> 10;
  const float* base = xh + (size_t)b * SEQc * D_MODELc + d;
  float s = 0.f;
  for (int l = 0; l < SEQc; ++l) s += base[(size_t)l * D_MODELc];
  const float mn = s * (1.f / SEQc);
  for (int l = 0; l < SEQc; ++l)
    out[(size_t)(b*SEQc + l) * D_MODELc + d] = base[(size_t)l * D_MODELc] - mn;
}

// trend conv: (B,L,1024) x trend_w(7,1024,3) -> += trend_acc(B,L,7), circular
__global__ __launch_bounds__(256) void trendconv_kernel(const float* __restrict__ t,
                                                        const float* __restrict__ w,
                                                        float* __restrict__ acc)
{
  __shared__ float red[256];
  const int bl = blockIdx.x;           // b*SEQ + l
  const int l = bl % SEQc, b = bl / SEQc;
  const int tid = threadIdx.x;
  const int lm = (l + SEQc - 1) % SEQc, lp = (l + 1) % SEQc;
  const float* t0 = t + (size_t)(b*SEQc + lm) * D_MODELc;
  const float* t1 = t + (size_t)(b*SEQc + l ) * D_MODELc;
  const float* t2 = t + (size_t)(b*SEQc + lp) * D_MODELc;
  for (int o = 0; o < NFEATc; ++o) {
    float s = 0.f;
    const float* wo = w + (size_t)o * D_MODELc * 3;
    for (int c = tid; c < D_MODELc; c += 256)
      s += t0[c]*wo[c*3+0] + t1[c]*wo[c*3+1] + t2[c]*wo[c*3+2];
    red[tid] = s; __syncthreads();
    for (int st = 128; st > 0; st >>= 1) {
      if (tid < st) red[tid] += red[tid+st];
      __syncthreads();
    }
    if (tid == 0) acc[(size_t)bl * NFEATc + o] += red[0];
    __syncthreads();
  }
}

// decoder input prep: trend_acc init (trend_init tail | seq-mean), d_cat
__global__ void decprep_kernel(const float* __restrict__ x_enc, const float* __restrict__ x_dec,
                               const float* __restrict__ s_init, const float* __restrict__ t_init,
                               float* __restrict__ trend_acc, float* __restrict__ d_cat)
{
  const int i = blockIdx.x * blockDim.x + threadIdx.x;
  if (i >= BATCHc * SEQc * NFEATc) return;
  const int c = i % NFEATc;
  const int l = (i / NFEATc) % SEQc;
  const int b = i / (NFEATc * SEQc);
  float tr;
  if (l < HALFSEQc) {
    tr = t_init[(size_t)(b*SEQc + HALFSEQc + l)*NFEATc + c];
  } else {
    float s = 0.f;
    for (int j = 0; j < SEQc; ++j) s += x_enc[(size_t)(b*SEQc + j)*NFEATc + c];
    tr = s * (1.f / SEQc);
  }
  trend_acc[i] = tr;
  d_cat[i] = (l < HALFSEQc) ? s_init[(size_t)(b*SEQc + HALFSEQc + l)*NFEATc + c]
                            : x_dec[(size_t)(b*SEQc + l)*NFEATc + c];
}

__global__ void markcat_kernel(const float* __restrict__ m_enc, const float* __restrict__ m_dec,
                               float* __restrict__ out)
{
  const int i = blockIdx.x * blockDim.x + threadIdx.x;
  if (i >= BATCHc * SEQc * NMARKc) return;
  const int m = i % NMARKc;
  const int l = (i / NMARKc) % SEQc;
  const int b = i / (NMARKc * SEQc);
  out[i] = (l < HALFSEQc) ? m_enc[(size_t)(b*SEQc + HALFSEQc + l)*NMARKc + m]
                          : m_dec[(size_t)(b*SEQc + l)*NMARKc + m];
}

// final: out = trend_acc + d @ proj_w + proj_b
__global__ void final_kernel(const float* __restrict__ d, const float* __restrict__ trend_acc,
                             const float* __restrict__ pw, const float* __restrict__ pb,
                             float* __restrict__ out)
{
  const int i = blockIdx.x * blockDim.x + threadIdx.x;
  if (i >= BATCHc * SEQc * NFEATc) return;
  const int o = i % NFEATc;
  const int row = i / NFEATc;
  const float* dr = d + (size_t)row * D_MODELc;
  float s = pb[o];
  for (int k = 0; k < D_MODELc; ++k) s += dr[k] * pw[k*NFEATc + o];
  out[i] = trend_acc[i] + s;
}

// ---------------------------------------------------------------------------
// Host orchestration
// ---------------------------------------------------------------------------
extern "C" void kernel_launch(void* const* d_in, const int* in_sizes, int n_in,
                              void* d_out, int out_size, void* d_ws, size_t ws_size,
                              hipStream_t stream)
{
  auto IN = [&](int i) -> const float* { return (const float*)d_in[i]; };
  // input index map (insertion-order flatten of setup_inputs())
  // 0 x_enc, 1 x_mark_enc, 2 x_dec, 3 x_mark_dec, 4 t,
  // 5 enc_emb.conv_w, 6 enc_emb.temp_w, 7 dec_emb.conv_w, 8 dec_emb.temp_w,
  // 9..58  enc layers (10 each: wq,bq,wk,bk,wv,bv,wo,bo,w1,w2)
  // 59..324 dec layers (19 each: self[8], cross[8], w1, w2, trend_w)
  // 325 enc_norm.g, 326 enc_norm.b, 327 dec_norm.g, 328 dec_norm.b, 329 proj.w, 330 proj.b

  char* wsb = (char*)d_ws;
  size_t off = 0;
  auto allocf = [&](size_t n) -> float* {
    float* p = (float*)(wsb + off);
    off += ((n * sizeof(float) + 255) / 256) * 256;
    return p;
  };
  float* A_h   = allocf((size_t)BLc * D_MODELc);
  float* A_enc = allocf((size_t)BLc * D_MODELc);
  float* Tq    = allocf((size_t)BLc * D_MODELc);
  float* Tk    = allocf((size_t)BLc * D_MODELc);
  float* Tv    = allocf((size_t)BLc * D_MODELc);
  float* Ttmp  = allocf((size_t)BLc * D_MODELc);
  float* Tts   = allocf((size_t)BLc * D_MODELc);
  float* Tffn  = allocf((size_t)BLc * D_FFc);
  float* corr  = allocf((size_t)BATCHc * SEQc);
  int*   delay = (int*)allocf((size_t)BATCHc * 16);
  float* tcw   = allocf((size_t)BATCHc * 16);
  float* s_init    = allocf((size_t)BATCHc * SEQc * NFEATc);
  float* t_init    = allocf((size_t)BATCHc * SEQc * NFEATc);
  float* d_cat     = allocf((size_t)BATCHc * SEQc * NFEATc);
  float* mark_cat  = allocf((size_t)BATCHc * SEQc * NMARKc);
  float* trend_acc = allocf((size_t)BATCHc * SEQc * NFEATc);
  (void)ws_size; (void)n_in; (void)in_sizes; (void)out_size;

  auto gemm = [&](const float* A, const float* Bw, const float* bias,
                  const float* resid, float* C, int M, int N, int K, int act) {
    dim3 grid(N / TN, M / TM);
    gemm_wmma_kernel<<<grid, 256, 0, stream>>>(A, Bw, bias, resid, C, M, N, K, act);
  };
  auto decomp = [&](const float* x, float* seas, float* trend, int C, int mode) {
    int tot = BATCHc * SEQc * C;
    decomp_kernel<<<(tot + 255) / 256, 256, 0, stream>>>(x, seas, trend, C, mode);
  };
  auto attn = [&](const float* xq, const float* xkv, int base, float* outbuf) {
    gemm(xq,  IN(base+0), IN(base+1), nullptr, Tq, BLc, D_MODELc, D_MODELc, 0);
    gemm(xkv, IN(base+2), IN(base+3), nullptr, Tk, BLc, D_MODELc, D_MODELc, 0);
    gemm(xkv, IN(base+4), IN(base+5), nullptr, Tv, BLc, D_MODELc, D_MODELc, 0);
    corr_kernel<<<dim3(SEQc, BATCHc), 256, 0, stream>>>(Tq, Tk, corr);
    topk_kernel<<<BATCHc, 64, 0, stream>>>(corr, delay, tcw);
    gather_kernel<<<(BLc * D_MODELc) / 256, 256, 0, stream>>>(Tv, delay, tcw, Tq);
    gemm(Tq, IN(base+6), IN(base+7), xq, outbuf, BLc, D_MODELc, D_MODELc, 0);
  };

  // ---------------- encoder ----------------
  embed_kernel<<<(BLc * D_MODELc) / 256, 256, 0, stream>>>(IN(0), IN(1), IN(5), IN(6), A_h);
  for (int li = 0; li < 5; ++li) {
    const int base = 9 + 10 * li;
    attn(A_h, A_h, base, Ttmp);
    decomp(Ttmp, A_h, nullptr, D_MODELc, 0);
    gemm(A_h,  IN(base+8), nullptr, nullptr, Tffn, BLc, D_FFc, D_MODELc, 1);   // gelu(x@w1)
    gemm(Tffn, IN(base+9), nullptr, A_h,    Ttmp, BLc, D_MODELc, D_FFc,   0);  // x + y
    decomp(Ttmp, A_h, nullptr, D_MODELc, 0);
  }
  ln1_kernel<<<BLc, 256, 0, stream>>>(A_h, Tk, IN(325), IN(326));
  ln2_kernel<<<(BATCHc * D_MODELc) / 256, 256, 0, stream>>>(Tk, A_enc);

  // ---------------- decoder prep ----------------
  decomp(IN(0), s_init, t_init, NFEATc, 1);
  decprep_kernel<<<(BATCHc*SEQc*NFEATc + 255)/256, 256, 0, stream>>>(
      IN(0), IN(2), s_init, t_init, trend_acc, d_cat);
  markcat_kernel<<<(BATCHc*SEQc*NMARKc + 255)/256, 256, 0, stream>>>(IN(1), IN(3), mark_cat);
  embed_kernel<<<(BLc * D_MODELc) / 256, 256, 0, stream>>>(d_cat, mark_cat, IN(7), IN(8), A_h);

  // ---------------- decoder ----------------
  for (int li = 0; li < 14; ++li) {
    const int base = 59 + 19 * li;
    attn(A_h, A_h, base + 0, Ttmp);                 // self
    decomp(Ttmp, A_h, Tts, D_MODELc, 1);            // t1 -> store
    attn(A_h, A_enc, base + 8, Ttmp);               // cross
    decomp(Ttmp, A_h, Tts, D_MODELc, 2);            // t2 -> accumulate
    gemm(A_h,  IN(base+16), nullptr, nullptr, Tffn, BLc, D_FFc, D_MODELc, 1);
    gemm(Tffn, IN(base+17), nullptr, A_h,    Ttmp, BLc, D_MODELc, D_FFc,   0);
    decomp(Ttmp, A_h, Tts, D_MODELc, 2);            // t3 -> accumulate
    trendconv_kernel<<<BLc, 256, 0, stream>>>(Tts, IN(base+18), trend_acc);
  }
  ln1_kernel<<<BLc, 256, 0, stream>>>(A_h, Tk, IN(327), IN(328));
  ln2_kernel<<<(BATCHc * D_MODELc) / 256, 256, 0, stream>>>(Tk, Ttmp);

  final_kernel<<<(BATCHc*SEQc*NFEATc + 255)/256, 256, 0, stream>>>(
      Ttmp, trend_acc, IN(329), IN(330), (float*)d_out);
}
